// H2GCN_Original_74998718922882
// MI455X (gfx1250) — matrix operand order repaired and checked by reading, not compiled, not run on hardware
//
#include <hip/hip_runtime.h>
#include <hip/hip_bf16.h>

// ---------------------------------------------------------------------------
// H2GCN on MI455X (gfx1250).  All large matmuls (A@A, hop aggregations, embed)
// run through v_wmma_f32_16x16x32_bf16 with LDS-staged operands:
//   - A panel moved global->LDS by the Tensor Data Mover (tensor_load_to_lds,
//     TENSORcnt) with hardware row padding to the 80B LDS pitch; manual
//     b128 staging as fallback when the builtin is unavailable.
//   - B panel staged TRANSPOSED in LDS so both WMMA fragments are two
//     contiguous ds_load_b128 per lane.
//   - 8 waves/block, each wave owns a 32x32 register tile (4 wmma / K-chunk).
// bf16 is exact for the binary adjacency, so (A@A - A - I) > 0 is computed
// exactly with a fused epilogue.  Adjacency (32MB bf16) is L2-resident.
// ---------------------------------------------------------------------------

#define NN        4096
#define IND       1024
#define HID       128
#define NCLS      10
#define FINAL_DIM 896   // (2^(K+1)-1)*HIDDEN, K=2

typedef __attribute__((ext_vector_type(16))) __bf16        v16bf;
typedef __attribute__((ext_vector_type(8)))  float         v8f;
typedef __attribute__((ext_vector_type(4)))  unsigned int  u32x4;
typedef __attribute__((ext_vector_type(4)))  int           i32x4;
typedef __attribute__((ext_vector_type(8)))  int           i32x8;

union BFrag { v16bf v; u32x4 q[2]; };

#if __has_builtin(__builtin_amdgcn_tensor_load_to_lds) && \
    __has_builtin(__builtin_amdgcn_s_wait_tensorcnt)
#define USE_TDM 1
#else
#define USE_TDM 0
#endif

__device__ __forceinline__ unsigned short f2bf(float f) {
    union { float f; unsigned int u; } c; c.f = f;
    unsigned int u = c.u;
    u += 0x7FFFu + ((u >> 16) & 1u);          // round-to-nearest-even
    return (unsigned short)(u >> 16);
}
__device__ __forceinline__ float bf2f(unsigned short h) {
    union { unsigned int u; float f; } c; c.u = ((unsigned int)h) << 16;
    return c.f;
}

// -------------------------- elementwise helpers ----------------------------

__global__ void scatter_edges_kernel(const int* __restrict__ edges,
                                     unsigned short* __restrict__ A,
                                     int nEdges) {
    int e = blockIdx.x * blockDim.x + threadIdx.x;
    if (e >= nEdges) return;
    int src = edges[e];
    int dst = edges[nEdges + e];
    A[(size_t)src * NN + dst] = 0x3F80;       // bf16 1.0 (races store same value)
}

__global__ void cvt_f32_bf16_kernel(const float* __restrict__ in,
                                    unsigned short* __restrict__ out, int count) {
    int i = blockIdx.x * blockDim.x + threadIdx.x;
    if (i < count) out[i] = f2bf(in[i]);
}

// one block (256 threads) per row: dinv[row] = rsqrt(sum(A[row,:]) + 1e-8)
__global__ void row_deg_rsqrt_kernel(const unsigned short* __restrict__ A,
                                     float* __restrict__ dinv) {
    __shared__ float red[256];
    int row = blockIdx.x;
    const unsigned short* r = A + (size_t)row * NN;
    float s = 0.f;
    for (int j = threadIdx.x; j < NN; j += 256) s += bf2f(r[j]);
    red[threadIdx.x] = s;
    __syncthreads();
    for (int off = 128; off > 0; off >>= 1) {
        if (threadIdx.x < off) red[threadIdx.x] += red[threadIdx.x + off];
        __syncthreads();
    }
    if (threadIdx.x == 0) dinv[row] = rsqrtf(red[0] + 1e-8f);
}

// A[i,j] *= dinv[i]*dinv[j]   (in place, bf16)
__global__ void scale_sym_kernel(unsigned short* __restrict__ A,
                                 const float* __restrict__ dinv) {
    size_t idx = (size_t)blockIdx.x * blockDim.x + threadIdx.x;
    if (idx >= (size_t)NN * NN) return;
    int i = (int)(idx >> 12);                  // /4096
    int j = (int)(idx & 4095);
    unsigned short h = A[idx];
    if (h) A[idx] = f2bf(bf2f(h) * dinv[i] * dinv[j]);
}

// ------------------------ LDS-staged WMMA GEMM -----------------------------
// C[4096 x w] = Aop[4096 x K; lda] @ Bop[K x w; ldb]
// grid = (4096/128, w/64), block = 256 threads (8 waves, 4x2 wave grid),
// each wave computes a 32x32 register tile (2x2 WMMA 16x16 tiles).
// MODE 0: store bf16.  MODE 1: A2 mask ((C - Araw - I) > 0).  MODE 2: bias+relu.

#define LDSPITCH 40   // u16 per LDS row (80B = 5*16B: aligned + conflict-free)

template <int MODE>
__global__ void gemm_bf16_kernel(const unsigned short* __restrict__ Aop, int lda,
                                 const unsigned short* __restrict__ Bop, int ldb,
                                 unsigned short* __restrict__ Cop, int ldc,
                                 int K,
                                 const unsigned short* __restrict__ Araw,
                                 const float* __restrict__ bias) {
    __shared__ __align__(16) unsigned short As[128][LDSPITCH];   // A panel, row-major
    __shared__ __align__(16) unsigned short BTs[64][LDSPITCH];   // B panel, TRANSPOSED [n][k]

    const int tid    = threadIdx.x;
    const int lane   = tid & 31;
    const int wave   = tid >> 5;
    const int wm     = wave >> 1;          // 0..3  (M super-tile)
    const int wn     = wave & 1;           // 0..1  (N super-tile)
    const int half   = lane >> 4;
    const int mn     = lane & 15;
    const int blockM = blockIdx.x * 128;
    const int blockN = blockIdx.y * 64;

    // staging maps
    const int bk = tid >> 3;               // 0..31
    const int bn = (tid & 7) * 8;          // 0..56
#if !USE_TDM
    const int arow0 = tid >> 2;            // rows 0..63
    const int apart = (tid & 3) * 8;       // u16 offset within 32-wide row
    const int arow1 = arow0 + 64;          // rows 64..127
#endif

#if USE_TDM
    // Tensor DMA descriptor (D#), invariant parts.  A-panel tile:
    // 128 rows x 64B, data_size=8B, row stride = lda*2 bytes, LDS pitch 80B
    // via pad_enable (interval code 3 = 16 dwords, amount code 3 = 4 dwords).
    const unsigned ldsAs = (unsigned)(size_t)(void*)&As[0][0];   // LDS byte offset
    i32x8 g1 = {0, 0, 0, 0, 0, 0, 0, 0};
    g1[0] = (3 << 16)            // data_size = 8B
          | (1 << 20)            // pad_enable
          | (3 << 22)            // pad_interval: 16 dwords (one 64B row)
          | (3 << 25);           // pad_amount:   4 dwords (16B)
    g1[1] = 0;                                   // tensor_dim0 low16 -> d1[31:16]=0
    g1[2] = 16 | (0 << 16);                      // tensor_dim0 = 1<<20 (hi), dim1 lo16=0
    g1[3] = 16 | (8 << 16);                      // tensor_dim1 = 1<<20 (hi), tile_dim0 = 8
    g1[4] = 128;                                 // tile_dim1 = 128 rows (tile_dim2 = 0)
    g1[5] = (int)((size_t)lda >> 2);             // tensor_dim0_stride = lda*2/8 units
    g1[6] = 0;
    g1[7] = 0;
    i32x4 gz4 = {0, 0, 0, 0};                    // groups 2/3 unused (2D tile)
    i32x8 gz8 = {0, 0, 0, 0, 0, 0, 0, 0};        // extra operand (clang-23 form)
#endif

    v8f acc[2][2] = {};

    for (int k0 = 0; k0 < K; k0 += 32) {
        __syncthreads();        // all waves done reading previous panels
        // ---- stage B panel transposed: BTs[n][k] ----
        union { u32x4 q; unsigned short s[8]; } t;
        t.q = *(const u32x4*)(Bop + (size_t)(k0 + bk) * ldb + blockN + bn);
#if USE_TDM
        if (tid == 0) {         // one DMA per chunk; EXEC-ignored, wave-scoped
            size_t ga = (size_t)(const void*)(Aop + (size_t)blockM * lda + k0);
            u32x4 g0;
            g0[0] = 1u;                                        // count = 1
            g0[1] = ldsAs;                                     // lds_addr
            g0[2] = (unsigned)(ga & 0xFFFFFFFFu);              // global_addr lo
            g0[3] = (unsigned)((ga >> 32) & 0x01FFFFFFu)       // global_addr hi
                  | 0x80000000u;                               // type = 2 ("image")
            __builtin_amdgcn_tensor_load_to_lds(g0, g1, gz4, gz4, gz8, 0);
            __builtin_amdgcn_s_wait_tensorcnt(0);
        }
#else
        // ---- fallback: manual A panel staging, loads before stores ----
        u32x4 va0 = *(const u32x4*)(Aop + (size_t)(blockM + arow0) * lda + k0 + apart);
        u32x4 va1 = *(const u32x4*)(Aop + (size_t)(blockM + arow1) * lda + k0 + apart);
        *(u32x4*)&As[arow0][apart] = va0;
        *(u32x4*)&As[arow1][apart] = va1;
#endif
        #pragma unroll
        for (int i = 0; i < 8; ++i) BTs[bn + i][bk] = t.s[i];
        if (k0 + 32 < K) {      // prefetch next B panel (global_prefetch_b8)
            __builtin_prefetch(Bop + (size_t)(k0 + 32 + bk) * ldb + blockN + bn, 0, 1);
#if !USE_TDM
            __builtin_prefetch(Aop + (size_t)(blockM + arow0) * lda + k0 + 32 + apart, 0, 1);
#endif
        }
        __syncthreads();        // panels published (DMA waited + DS waited)

        // ---- fragments: 2 contiguous b128 LDS loads each ----
        BFrag a[2], b[2];
        #pragma unroll
        for (int mt = 0; mt < 2; ++mt) {
            const unsigned short* r = &As[wm * 32 + mt * 16 + mn][0];
            a[mt].q[0] = *(const u32x4*)(r + 8 * half);        // K 0..7  (+8*half)
            a[mt].q[1] = *(const u32x4*)(r + 16 + 8 * half);   // K 16..23(+8*half)
        }
        #pragma unroll
        for (int nt = 0; nt < 2; ++nt) {
            const unsigned short* r = &BTs[wn * 32 + nt * 16 + mn][0];
            b[nt].q[0] = *(const u32x4*)(r + 8 * half);        // K 0..7  (+8*half)
            b[nt].q[1] = *(const u32x4*)(r + 16 + 8 * half);   // K 16..23(+8*half)
        }
        #pragma unroll
        for (int mt = 0; mt < 2; ++mt)
            #pragma unroll
            for (int nt = 0; nt < 2; ++nt)
                acc[mt][nt] = __builtin_amdgcn_wmma_f32_16x16x32_bf16(
                    false, a[mt].v, false, b[nt].v, (short)0, acc[mt][nt],
                    false, false);
    }

    // ---- epilogue ----
    const int mOff = half * 8;
    #pragma unroll
    for (int mt = 0; mt < 2; ++mt) {
        #pragma unroll
        for (int nt = 0; nt < 2; ++nt) {
            #pragma unroll
            for (int v = 0; v < 8; ++v) {
                int gm = blockM + wm * 32 + mt * 16 + mOff + v;
                int gn = blockN + wn * 32 + nt * 16 + mn;
                float c = acc[mt][nt][v];
                if (MODE == 1) {
                    float sub = bf2f(Araw[(size_t)gm * NN + gn]) + (gm == gn ? 1.f : 0.f);
                    Cop[(size_t)gm * ldc + gn] = (c - sub > 0.f) ? 0x3F80 : 0;
                } else if (MODE == 2) {
                    c += bias[gn];
                    c = c > 0.f ? c : 0.f;
                    Cop[(size_t)gm * ldc + gn] = f2bf(c);
                } else {
                    Cop[(size_t)gm * ldc + gn] = f2bf(c);
                }
            }
        }
    }
}

// ---- classifier: out[4096,10] = Rf @ W_cls + b_cls  (f32) -----------------
__global__ void classify_kernel(const unsigned short* __restrict__ Rf,
                                const float* __restrict__ Wc,
                                const float* __restrict__ bc,
                                float* __restrict__ out) {
    int idx = blockIdx.x * blockDim.x + threadIdx.x;
    if (idx >= NN * NCLS) return;
    int row = idx / NCLS;
    int cls = idx - row * NCLS;
    const unsigned short* r = Rf + (size_t)row * FINAL_DIM;
    float s = bc[cls];
    for (int k = 0; k < FINAL_DIM; ++k)
        s += bf2f(r[k]) * Wc[k * NCLS + cls];
    out[idx] = s;
}

// ---------------------------------------------------------------------------

extern "C" void kernel_launch(void* const* d_in, const int* in_sizes, int n_in,
                              void* d_out, int out_size, void* d_ws, size_t ws_size,
                              hipStream_t stream) {
    const float* X     = (const float*)d_in[0];
    const int*   edges = (const int*)  d_in[1];
    const float* Wemb  = (const float*)d_in[2];
    const float* bemb  = (const float*)d_in[3];
    const float* Wcls  = (const float*)d_in[4];
    const float* bcls  = (const float*)d_in[5];
    float* out = (float*)d_out;
    int nEdges = in_sizes[1] / 2;

    // workspace carve-up (~83 MB)
    char* ws = (char*)d_ws;
    unsigned short* A   = (unsigned short*)ws; ws += (size_t)NN * NN * 2;        // 32 MB
    unsigned short* A2  = (unsigned short*)ws; ws += (size_t)NN * NN * 2;        // 32 MB
    unsigned short* Xbf = (unsigned short*)ws; ws += (size_t)NN * IND * 2;       // 8 MB
    unsigned short* Wbf = (unsigned short*)ws; ws += (size_t)IND * HID * 2;      // 256 KB
    unsigned short* Rf  = (unsigned short*)ws; ws += (size_t)NN * FINAL_DIM * 2; // 7 MB
    float* d1 = (float*)ws; ws += (size_t)NN * 4;
    float* d2 = (float*)ws; ws += (size_t)NN * 4;

    // 1) adjacency (bf16 0/1)
    (void)hipMemsetAsync(A, 0, (size_t)NN * NN * 2, stream);
    scatter_edges_kernel<<<(nEdges + 255) / 256, 256, 0, stream>>>(edges, A, nEdges);

    // 2) convert features / weights to bf16
    cvt_f32_bf16_kernel<<<(NN * IND + 255) / 256, 256, 0, stream>>>(X, Xbf, NN * IND);
    cvt_f32_bf16_kernel<<<(IND * HID + 255) / 256, 256, 0, stream>>>(Wemb, Wbf, IND * HID);

    // 3) deg(A) -> d1
    row_deg_rsqrt_kernel<<<NN, 256, 0, stream>>>(A, d1);

    // 4) A2 = ((A@A) - A - I) > 0   (exact in bf16 WMMA; reads raw A)
    gemm_bf16_kernel<1><<<dim3(NN / 128, NN / 64), 256, 0, stream>>>(
        A, NN, A, NN, A2, NN, NN, A, nullptr);

    // 5) deg(A2) -> d2; symmetric normalization in place
    row_deg_rsqrt_kernel<<<NN, 256, 0, stream>>>(A2, d2);
    scale_sym_kernel<<<(int)(((size_t)NN * NN + 255) / 256), 256, 0, stream>>>(A, d1);
    scale_sym_kernel<<<(int)(((size_t)NN * NN + 255) / 256), 256, 0, stream>>>(A2, d2);

    // 6) r0 = relu(X @ W + b)  -> Rf[:, 0:128]
    gemm_bf16_kernel<2><<<dim3(NN / 128, HID / 64), 256, 0, stream>>>(
        Xbf, IND, Wbf, HID, Rf, FINAL_DIM, IND, nullptr, bemb);

    // 7) hop 1: A1@r0 -> cols 128:256, A2@r0 -> cols 256:384
    gemm_bf16_kernel<0><<<dim3(NN / 128, 128 / 64), 256, 0, stream>>>(
        A, NN, Rf + 0, FINAL_DIM, Rf + 128, FINAL_DIM, NN, nullptr, nullptr);
    gemm_bf16_kernel<0><<<dim3(NN / 128, 128 / 64), 256, 0, stream>>>(
        A2, NN, Rf + 0, FINAL_DIM, Rf + 256, FINAL_DIM, NN, nullptr, nullptr);

    //    hop 2: input = cols 128:384 (w=256): A1@ -> 384:640, A2@ -> 640:896
    gemm_bf16_kernel<0><<<dim3(NN / 128, 256 / 64), 256, 0, stream>>>(
        A, NN, Rf + 128, FINAL_DIM, Rf + 384, FINAL_DIM, NN, nullptr, nullptr);
    gemm_bf16_kernel<0><<<dim3(NN / 128, 256 / 64), 256, 0, stream>>>(
        A2, NN, Rf + 128, FINAL_DIM, Rf + 640, FINAL_DIM, NN, nullptr, nullptr);

    // 8) out = Rf @ W_cls + b_cls
    classify_kernel<<<(NN * NCLS + 255) / 256, 256, 0, stream>>>(Rf, Wcls, bcls, out);
}
